// JointWiseFeedForward_29283087024530
// MI455X (gfx1250) — compile-verified, matrix operand order checked by reading, not run on hardware
//
#include <hip/hip_runtime.h>
#include <hip/hip_bf16.h>

// ---------------------------------------------------------------------------
// JointWiseFeedForward (locally-connected per-token FFN), MI455X / gfx1250
//   per token t (16): H = gelu(X_t @ W1_t^T);  Y_t = H @ W2_t^T
//   X_t: [16384 x 128] (stride-16 interleaved in x), W1_t: [512 x 128],
//   W2_t: [128 x 512], output interleaved the same way.
// bf16 WMMA (v_wmma_f32_16x16x32_bf16) with f32 accumulation.
// GELU via hardware exp2/rcp (tanh form) to keep VALU off the critical path.
// ---------------------------------------------------------------------------

#define T_TOK    16
#define CIN      128
#define CFF      512
#define COUT     128
#define FDIM     2048          // = CIN * T_TOK = COUT * T_TOK
#define BM       128           // rows per workgroup
#define CHUNK    128           // ff channels per chunk
#define NCHUNK   (CFF / CHUNK) // 4
#define LDSW     136           // padded LDS row stride in bf16 elems (272 B, 16B-aligned)
#define NTHREADS 256           // 8 waves (wave32)

typedef __bf16 bf16_t;
typedef bf16_t v16bf __attribute__((ext_vector_type(16)));
typedef float  v8f   __attribute__((ext_vector_type(8)));

__device__ __forceinline__ unsigned int f32_to_bf16_bits(float f) {
  unsigned int u = __float_as_uint(f);
  u += 0x7fffu + ((u >> 16) & 1u);          // round-to-nearest-even
  return u >> 16;
}

__device__ __forceinline__ unsigned int pack_bf16x2(float lo, float hi) {
  return f32_to_bf16_bits(lo) | (f32_to_bf16_bits(hi) << 16);
}

// tanh-form GELU on hardware transcendentals:
//   gelu(x) = x * (1 - 1/(exp2(u)+1)),  u = x*(C0 + C1*x^2)
//   C0 = 2*log2(e)*0.7978845608, C1 = C0*0.044715
__device__ __forceinline__ float gelu_fast(float x) {
  const float C0 = 2.3022082f;
  const float C1 = 0.1029432f;
  float x2 = x * x;
  float u  = x * __builtin_fmaf(x2, C1, C0);
  float e  = __builtin_amdgcn_exp2f(u);          // v_exp_f32
  float r  = __builtin_amdgcn_rcpf(e + 1.0f);    // v_rcp_f32
  return __builtin_fmaf(-x, r, x);               // x - x*r
}

union OpU {
  uint4 q[2];
  v16bf v;
};

// Load one 16x32 bf16 WMMA operand from an LDS tile stored [row][k] row-major.
// CDNA5 16-bit operand striping: lane(0..15) -> row, elements 0..7 = K(kk..kk+7),
// elements 8..15 = K(kk+16..kk+23); lanes 16..31 take the +8 K slices.
__device__ __forceinline__ v16bf load_op(const unsigned short* __restrict__ tile,
                                         int row0, int kk, int lane) {
  const int r  = row0 + (lane & 15);
  const int hi = (lane >> 4) << 3;          // 0 or 8
  const unsigned short* p = tile + r * LDSW + kk + hi;
  OpU u;
  u.q[0] = *(const uint4*)(p);              // ds_load_b128
  u.q[1] = *(const uint4*)(p + 16);         // ds_load_b128
  return u.v;
}

__global__ void __launch_bounds__(NTHREADS)
jointwise_ffn_wmma(const float* __restrict__ x,
                   const float* __restrict__ w1,
                   const float* __restrict__ w2,
                   float* __restrict__ out) {
  extern __shared__ unsigned short smem[];
  unsigned short* sA  = smem;                 // [BM][LDSW]    X tile (bf16)
  unsigned short* sW1 = sA  + BM    * LDSW;   // [CHUNK][LDSW] W1 chunk [o_local][c]
  unsigned short* sW2 = sW1 + CHUNK * LDSW;   // [COUT][LDSW]  W2 chunk [o][ff_local]
  unsigned short* sH  = sW2 + COUT  * LDSW;   // [BM][LDSW]    gelu(H) chunk (bf16)

  const int tid  = threadIdx.x;
  const int lane = tid & 31;
  const int wv   = tid >> 5;                        // wave 0..7
  const int t    = blockIdx.x & (T_TOK - 1);        // token id
  const int m0   = (blockIdx.x >> 4) * BM;          // row-block base
  const int rowstrip = wv * 16;                     // this wave's 16-row strip

  // ---- stage X tile (strided token gather) as bf16 into LDS, 2 elems/iter ----
  for (int i = tid; i < (BM * CIN) / 2; i += NTHREADS) {
    const int idx = i << 1;
    const int r = idx >> 7, c = idx & (CIN - 1);    // c even
    const float* px = x + (size_t)(m0 + r) * FDIM + c * T_TOK + t;
    *(unsigned int*)(sA + r * LDSW + c) = pack_bf16x2(px[0], px[T_TOK]);
  }

  const v8f vzero = {0.f, 0.f, 0.f, 0.f, 0.f, 0.f, 0.f, 0.f};
  v8f yacc[8];
#pragma unroll
  for (int n = 0; n < 8; ++n) yacc[n] = vzero;

  for (int ffc = 0; ffc < NCHUNK; ++ffc) {
    // ---- stage W1 chunk: rows = local ff channel o, cols = k (cin) ----
    for (int i = tid; i < (CHUNK * CIN) / 2; i += NTHREADS) {
      const int idx = i << 1;
      const int o = idx >> 7, c = idx & (CIN - 1);
      const float* pw = w1 + (size_t)((ffc * CHUNK + o) * CIN + c) * T_TOK + t;
      *(unsigned int*)(sW1 + o * LDSW + c) = pack_bf16x2(pw[0], pw[T_TOK]);
    }
    // ---- stage W2 chunk: rows = out channel o, cols = k (local ff) ----
    for (int i = tid; i < (COUT * CHUNK) / 2; i += NTHREADS) {
      const int idx = i << 1;
      const int o = idx >> 7, c = idx & (CHUNK - 1);
      const float* pw = w2 + (size_t)(o * CFF + ffc * CHUNK + c) * T_TOK + t;
      *(unsigned int*)(sW2 + o * LDSW + c) = pack_bf16x2(pw[0], pw[T_TOK]);
    }
    // hint next chunk's weights into cache (global_prefetch_b8)
    if (ffc + 1 < NCHUNK) {
      __builtin_prefetch(&w1[(size_t)(((ffc + 1) * CHUNK + (tid >> 1)) * CIN) * T_TOK + t], 0, 0);
      __builtin_prefetch(&w2[(size_t)((tid >> 1) * CFF + (ffc + 1) * CHUNK) * T_TOK + t], 0, 0);
    }
    __syncthreads();

    // ---- GEMM1: H(strip) = X(strip) @ W1^T, K = CIN = 128 ----
    v16bf aop[4];
#pragma unroll
    for (int kk = 0; kk < 4; ++kk) aop[kk] = load_op(sA, rowstrip, kk * 32, lane);

    const int hrow0 = rowstrip + ((lane >> 4) << 3);
#pragma unroll
    for (int n = 0; n < 8; ++n) {
      v8f h = vzero;
#pragma unroll
      for (int kk = 0; kk < 4; ++kk) {
        const v16bf bop = load_op(sW1, n * 16, kk * 32, lane);
        h = __builtin_amdgcn_wmma_f32_16x16x32_bf16(false, aop[kk], false, bop,
                                                    (short)0, h, false, false);
      }
      // GELU in f32, write bf16 H to LDS (C layout: M = r + 8*(lane>=16))
      const int hcol = n * 16 + (lane & 15);
#pragma unroll
      for (int r = 0; r < 8; ++r)
        sH[(hrow0 + r) * LDSW + hcol] =
            (unsigned short)f32_to_bf16_bits(gelu_fast(h[r]));
    }
    __syncthreads();

    // ---- GEMM2: Y(strip) += H(strip) @ W2^T, K = CHUNK = 128 ----
    v16bf hop[4];
#pragma unroll
    for (int kk = 0; kk < 4; ++kk) hop[kk] = load_op(sH, rowstrip, kk * 32, lane);
#pragma unroll
    for (int n = 0; n < 8; ++n) {
#pragma unroll
      for (int kk = 0; kk < 4; ++kk) {
        const v16bf bop = load_op(sW2, n * 16, kk * 32, lane);
        yacc[n] = __builtin_amdgcn_wmma_f32_16x16x32_bf16(false, hop[kk], false, bop,
                                                          (short)0, yacc[n], false, false);
      }
    }
    __syncthreads();   // protect sW1/sW2/sH before next chunk overwrites
  }

  // ---- write Y strip (interleaved feature index o*16 + t), f32 ----
  const int lm = (lane >> 4) << 3;
  const int ln = lane & 15;
#pragma unroll
  for (int n = 0; n < 8; ++n) {
#pragma unroll
    for (int r = 0; r < 8; ++r) {
      const int row = m0 + rowstrip + lm + r;
      out[(size_t)row * FDIM + (n * 16 + ln) * T_TOK + t] = yacc[n][r];
    }
  }
}

extern "C" void kernel_launch(void* const* d_in, const int* in_sizes, int n_in,
                              void* d_out, int out_size, void* d_ws, size_t ws_size,
                              hipStream_t stream) {
  (void)n_in; (void)d_ws; (void)ws_size; (void)out_size;
  const float* x  = (const float*)d_in[0];   // [64,256,2048] f32
  const float* w1 = (const float*)d_in[1];   // [512,128,16]  f32
  const float* w2 = (const float*)d_in[2];   // [128,512,16]  f32
  float* out = (float*)d_out;                // [64,256,2048] f32

  const int nrow   = in_sizes[0] / FDIM;           // 16384
  const int nblk   = (nrow / BM) * T_TOK;          // 128 * 16 = 2048
  const size_t lds = (size_t)(BM + CHUNK + COUT + BM) * LDSW * sizeof(unsigned short); // 139264 B

  jointwise_ffn_wmma<<<nblk, NTHREADS, lds, stream>>>(x, w1, w2, out);
}
